// LSNN_2RNN_general_87050397155448
// MI455X (gfx1250) — matrix operand order repaired and checked by reading, not compiled, not run on hardware
//
#include <hip/hip_runtime.h>
#include <hip/hip_bf16.h>
#include <math.h>

// ---------------------------------------------------------------------------
// LSNN 3-layer recurrent spiking network, 196 sequential steps, batch 256.
// MI455X / gfx1250 (wave32) strategy:
//  * bf16 WMMA 16x16x32 for all big GEMMs (spikes are exactly 0/1 in bf16).
//  * Persistent kernel; 2 device-wide barriers per step (the minimum for the
//    s1->L2 and s2->L3 same-step dependencies). The output GEMM lags two
//    steps and s3 is triple-buffered, which removes the third barrier.
//  * Static wave->tile ownership (512 waves <-> 512 16x16 tiles per layer);
//    mem/b/spk neuron state stays in VGPRs for all 196 steps.
//  * All 8 waves of a block share one ntile -> the four 16x512 weight tiles
//    a block ever needs are converted fp32->bf16 and staged in LDS ONCE in
//    WMMA B-fragment order (16B chunks, lane-major => conflict-light
//    ds_load_b128). No weight traffic after the prologue.
//  * A (spike) fragments stream from L2 with 2-deep software pipelining.
//  * Grid barrier = arrive counter + separate release word (one L2 write per
//    phase instead of 64 polled RMWs).
// ---------------------------------------------------------------------------

typedef __attribute__((ext_vector_type(16))) __bf16 bf16x16;
typedef __attribute__((ext_vector_type(8)))  __bf16 bf16x8;
typedef __attribute__((ext_vector_type(8)))  float  f32x8;

#define NBLOCKS   64
#define TPB       256          // 8 waves per block -> 512 waves total
#define H         512
#define M_BATCH   256
#define T_STEPS   196
#define FULL_LEN  784
#define S_ELEMS   (M_BATCH * H)   // one spike buffer (elements)

__device__ __forceinline__ f32x8 wmma_bf16(bf16x16 a, bf16x16 b, f32x8 c) {
  return __builtin_amdgcn_wmma_f32_16x16x32_bf16(false, a, false, b,
                                                 (short)0, c, false, false);
}

union FragU { bf16x16 v; bf16x8 h[2]; };

// A-fragment loader (contiguous 16-bit 16x32 layout): lane l holds row m0+(l&15),
// elems 0..15 = K k+(l>>4)*16+{0..15}.
__device__ __forceinline__ void load_a(FragU& f, const __bf16* ap, int k) {
  f.h[0] = *(const bf16x8*)(ap + k);
  f.h[1] = *(const bf16x8*)(ap + k + 8);
}

// B tile staged in LDS as 16-byte chunks: chunk(kc,h,lane) at element offset
// (kc*64 + h*32 + lane)*8 holds W[n0+(lane&15)][kc*32+(lane>>4)*16+h*8 .. +7].
__device__ __forceinline__ void load_b_lds(FragU& f, const __bf16* bp, int kc) {
  f.h[0] = *(const bf16x8*)(bp + (kc * 64 +  0) * 8);
  f.h[1] = *(const bf16x8*)(bp + (kc * 64 + 32) * 8);
}

// GEMM tile: A from global (2-deep pipelined), B from LDS-staged tile.
__device__ __forceinline__ f32x8 gemm_tile_ldsB(const __bf16* __restrict__ A,
                                                const __bf16* Btile,
                                                int m0, int lane, f32x8 c) {
  const __bf16* ap = A + (size_t)(m0 + (lane & 15)) * H + ((lane >> 4) << 4);
  const __bf16* bp = Btile + lane * 8;
  FragU a0, a1, b0, b1;
  load_a(a0, ap, 0);
#pragma unroll
  for (int kc = 0; kc < 16; kc += 2) {
    load_a(a1, ap, (kc + 1) * 32);          // prefetch next A stage
    load_b_lds(b0, bp, kc);
    c = wmma_bf16(a0.v, b0.v, c);
    if (kc + 2 < 16) load_a(a0, ap, (kc + 2) * 32);
    load_b_lds(b1, bp, kc + 1);
    c = wmma_bf16(a1.v, b1.v, c);
  }
  return c;
}

// GEMM tile with B from global bf16 (used only for the small output GEMM).
__device__ __forceinline__ f32x8 gemm_tile_g(const __bf16* __restrict__ A,
                                             const __bf16* __restrict__ B,
                                             int m0, int n0, int lane, f32x8 c) {
  const __bf16* ap = A + (size_t)(m0 + (lane & 15)) * H + ((lane >> 4) << 4);
  const __bf16* bp = B + (size_t)(n0 + (lane & 15)) * H + ((lane >> 4) << 4);
  FragU a0, a1, b0, b1;
  load_a(a0, ap, 0);
  b0.v = *(const bf16x16*)(bp);
#pragma unroll
  for (int k = 0; k < H; k += 64) {
    load_a(a1, ap, k + 32);
    b1.v = *(const bf16x16*)(bp + k + 32);
    c = wmma_bf16(a0.v, b0.v, c);
    if (k + 64 < H) {
      load_a(a0, ap, k + 64);
      b0.v = *(const bf16x16*)(bp + k + 64);
    }
    c = wmma_bf16(a1.v, b1.v, c);
  }
  return c;
}

// Device-wide barrier: arrive counter + once-per-phase release word.
__device__ __forceinline__ void grid_barrier(unsigned* cnt, unsigned* rel,
                                             unsigned phase) {
  __threadfence();
  __syncthreads();
  if (threadIdx.x == 0) {
    const unsigned old =
        __hip_atomic_fetch_add(cnt, 1u, __ATOMIC_ACQ_REL, __HIP_MEMORY_SCOPE_AGENT);
    if (old == phase * (unsigned)NBLOCKS - 1u) {
      __hip_atomic_store(rel, phase, __ATOMIC_RELEASE, __HIP_MEMORY_SCOPE_AGENT);
    } else {
      while (__hip_atomic_load(rel, __ATOMIC_ACQUIRE, __HIP_MEMORY_SCOPE_AGENT) < phase) {
        __builtin_amdgcn_s_sleep(2);
      }
    }
  }
  __syncthreads();
}

// ---------------------------------------------------------------------------
// Prep: convert padded output weights to bf16, zero spike buffers, reset
// barrier state. Runs every launch (deterministic, graph-capture safe).
// ---------------------------------------------------------------------------
__global__ void snn_prep(const float* __restrict__ Whof,
                         __bf16* Who, __bf16* s1, __bf16* s2, __bf16* s3,
                         unsigned* cnt, unsigned* rel) {
  const size_t i  = (size_t)blockIdx.x * blockDim.x + threadIdx.x;
  const size_t st = (size_t)gridDim.x * blockDim.x;
  for (size_t j = i; j < (size_t)16 * H; j += st) {      // pad 10 rows -> 16
    size_t r = j >> 9, c = j & (H - 1);
    Who[j] = (r < 10) ? (__bf16)Whof[r * H + c] : (__bf16)0.0f;
  }
  for (size_t j = i; j < (size_t)3 * S_ELEMS; j += st) { // s3 is triple-buffered
    if (j < 2 * (size_t)S_ELEMS) { s1[j] = (__bf16)0.0f; s2[j] = (__bf16)0.0f; }
    s3[j] = (__bf16)0.0f;
  }
  if (i == 0) { *cnt = 0u; *rel = 0u; }
}

// ---------------------------------------------------------------------------
// Main persistent kernel.
// ---------------------------------------------------------------------------
__global__ __launch_bounds__(TPB, 1) void snn_main(
    const float* __restrict__ x,         // 256 x 784
    const float* __restrict__ Wi1,       // 512 x 4 (fp32, VALU path)
    const float* __restrict__ bi1, const float* __restrict__ bh1,
    const float* __restrict__ bi2, const float* __restrict__ bh2,
    const float* __restrict__ bi3, const float* __restrict__ bo,
    const float* __restrict__ tau1, const float* __restrict__ tau2,
    const float* __restrict__ tau3,
    const float* __restrict__ mask1, const float* __restrict__ mask2,
    const float* __restrict__ mask3,     // each 512 x 784
    const float* __restrict__ Wh1f, const float* __restrict__ Wi2f,
    const float* __restrict__ Wh2f, const float* __restrict__ Wi3f, // fp32 weights
    const __bf16* __restrict__ Who,      // 16 x 512 bf16 (padded)
    __bf16* s1buf, __bf16* s2buf,        // each 2 * 256*512
    __bf16* s3buf,                       // 3 * 256*512 (triple-buffered)
    unsigned* bar_cnt, unsigned* bar_rel,
    float* __restrict__ out)             // 256 x 10
{
  const int lane  = threadIdx.x & 31;
  const int w     = blockIdx.x * (TPB / 32) + (threadIdx.x >> 5); // 0..511
  const int m0    = (w & 15) << 4;
  const int n0    = (blockIdx.x >> 1) << 4;   // shared by all waves in block
  const int n     = n0 + (lane & 15);
  const int rbase = m0 + ((lane >> 4) << 3);

  // ---- stage this block's four weight tiles in LDS, bf16, fragment order --
  __shared__ __align__(32) __bf16 ldsW[4][8192];   // 4 x 16KB = 64KB
  {
    const float* srcs[4] = {Wh1f, Wi2f, Wh2f, Wi3f};
#pragma unroll
    for (int wsel = 0; wsel < 4; ++wsel) {
      const float* Wf = srcs[wsel];
      for (int q = threadIdx.x; q < 1024; q += TPB) {
        const int kc = q >> 6, hh = (q >> 5) & 1, l = q & 31;
        const float* s = Wf + (size_t)(n0 + (l & 15)) * H +
                         kc * 32 + ((l >> 4) << 4) + hh * 8;
        const float4 f0 = *(const float4*)(s);
        const float4 f1 = *(const float4*)(s + 4);
        bf16x8 v = { (__bf16)f0.x, (__bf16)f0.y, (__bf16)f0.z, (__bf16)f0.w,
                     (__bf16)f1.x, (__bf16)f1.y, (__bf16)f1.z, (__bf16)f1.w };
        *(bf16x8*)&ldsW[wsel][q * 8] = v;
      }
    }
    __syncthreads();
  }

  const float alpha  = expf(-1.0f / 20.0f);
  const float ialpha = 1.0f - alpha;
  const float ro1 = expf(-1.0f / tau1[n]); const float ir1 = 1.0f - ro1;
  const float ro2 = expf(-1.0f / tau2[n]); const float ir2 = 1.0f - ro2;
  const float ro3 = expf(-1.0f / tau3[n]); const float ir3 = 1.0f - ro3;
  const float bs1 = bi1[n] + bh1[n];
  const float bs2 = bi2[n] + bh2[n];
  const float bs3 = bi3[n];
  float wi[4];
#pragma unroll
  for (int j = 0; j < 4; ++j) wi[j] = Wi1[n * 4 + j];
  const float* mk1p = mask1 + (size_t)n * FULL_LEN;
  const float* mk2p = mask2 + (size_t)n * FULL_LEN;
  const float* mk3p = mask3 + (size_t)n * FULL_LEN;

  float mem1[8], mem2[8], mem3[8], bb1[8], bb2[8], bb3[8], sp1[8], sp2[8], sp3[8];
#pragma unroll
  for (int r = 0; r < 8; ++r) {
    mem1[r] = mem2[r] = mem3[r] = 0.0f;
    bb1[r] = bb2[r] = bb3[r] = 0.01f;        // b starts at b_j0
    sp1[r] = sp2[r] = sp3[r] = 0.0f;
  }

  f32x8 acc = {};                             // output accumulator (waves 0..15)
  const bool is_out_wave = (w < 16);
  const int om0 = w << 4;

  unsigned phase = 0;
  int cur = 0;                 // s1/s2 ping-pong
  int s3w_i = 0, s3o_i = 1;    // s3 write idx = t%3, out-read idx = (t+1)%3

  for (int t = 0; t < T_STEPS; ++t) {
    const __bf16* s1r = s1buf + (size_t)cur * S_ELEMS;
    const __bf16* s2r = s2buf + (size_t)cur * S_ELEMS;
    __bf16* s1w = s1buf + (size_t)(cur ^ 1) * S_ELEMS;
    __bf16* s2w = s2buf + (size_t)(cur ^ 1) * S_ELEMS;
    __bf16* s3w = s3buf + (size_t)s3w_i * S_ELEMS;
    const __bf16* s3o = s3buf + (size_t)s3o_i * S_ELEMS;

    // out += s3(t-2) @ Who^T  (t<2 reads zeroed buffers -> adds 0;
    // ordering guaranteed by the two barriers of step t-1)
    if (is_out_wave) acc = gemm_tile_g(s3o, Who, om0, 0, lane, acc);

    // ---------------- layer 1: h1 = x_t@Wi1^T + s1@Wh1^T + biases ----------
    f32x8 h1v = {};
    h1v = gemm_tile_ldsB(s1r, ldsW[0], m0, lane, h1v);
    const int   c0  = (t < 48) ? (t * 4) : 780;    // faithful column schedule
    const float mk1 = mk1p[t];
#pragma unroll
    for (int r = 0; r < 8; ++r) {
      const int m = rbase + r;
      const float4 xv = *(const float4*)(x + (size_t)m * FULL_LEN + c0);
      const float inp = h1v[r] + xv.x * wi[0] + xv.y * wi[1] +
                        xv.z * wi[2] + xv.w * wi[3] + bs1;
      bb1[r] = ro1 * bb1[r] + ir1 * sp1[r];
      const float Bth = 0.01f + 1.8f * bb1[r];
      const float nm  = mem1[r] * alpha + ialpha * inp - Bth * sp1[r];
      mem1[r] = (mk1 != 0.0f) ? nm : mem1[r];
      const float sp = ((mem1[r] - Bth) > 0.0f ? 1.0f : 0.0f) * mk1;
      sp1[r] = sp;
      s1w[(size_t)m * H + n] = (__bf16)sp;
    }
    grid_barrier(bar_cnt, bar_rel, ++phase);

    // ---------------- layer 2: h2 = s1_new@Wi2^T + s2@Wh2^T + biases -------
    f32x8 h2v = {};
    h2v = gemm_tile_ldsB(s1w, ldsW[1], m0, lane, h2v);
    h2v = gemm_tile_ldsB(s2r, ldsW[2], m0, lane, h2v);
    const float mk2 = mk2p[t];
#pragma unroll
    for (int r = 0; r < 8; ++r) {
      const int m = rbase + r;
      const float inp = h2v[r] + bs2;
      bb2[r] = ro2 * bb2[r] + ir2 * sp2[r];
      const float Bth = 0.01f + 1.8f * bb2[r];
      const float nm  = mem2[r] * alpha + ialpha * inp - Bth * sp2[r];
      mem2[r] = (mk2 != 0.0f) ? nm : mem2[r];
      const float sp = ((mem2[r] - Bth) > 0.0f ? 1.0f : 0.0f) * mk2;
      sp2[r] = sp;
      s2w[(size_t)m * H + n] = (__bf16)sp;
    }
    grid_barrier(bar_cnt, bar_rel, ++phase);

    // ---------------- layer 3: h3 = s2_new@Wi3^T + bias (no recurrence) ----
    // No trailing barrier: s3w is not read until >=2 barriers from now.
    f32x8 h3v = {};
    h3v = gemm_tile_ldsB(s2w, ldsW[3], m0, lane, h3v);
    const float mk3 = mk3p[t];
#pragma unroll
    for (int r = 0; r < 8; ++r) {
      const int m = rbase + r;
      const float inp = h3v[r] + bs3;
      bb3[r] = ro3 * bb3[r] + ir3 * sp3[r];
      const float Bth = 0.01f + 1.8f * bb3[r];
      const float nm  = mem3[r] * alpha + ialpha * inp - Bth * sp3[r];
      mem3[r] = (mk3 != 0.0f) ? nm : mem3[r];
      const float sp = ((mem3[r] - Bth) > 0.0f ? 1.0f : 0.0f) * mk3;
      sp3[r] = sp;
      s3w[(size_t)m * H + n] = (__bf16)sp;
    }

    cur ^= 1;
    s3w_i = (s3w_i == 2) ? 0 : s3w_i + 1;
    s3o_i = (s3o_i == 2) ? 0 : s3o_i + 1;
  }

  // One final barrier so s3(194) and s3(195) are globally visible, then add
  // their contributions (loop covered s3(0..193)) and write acc/T + b_out.
  grid_barrier(bar_cnt, bar_rel, ++phase);
  if (is_out_wave) {
    acc = gemm_tile_g(s3buf + (size_t)2 * S_ELEMS, Who, om0, 0, lane, acc); // 194%3
    acc = gemm_tile_g(s3buf + (size_t)0 * S_ELEMS, Who, om0, 0, lane, acc); // 195%3
    const int nn = lane & 15;
    if (nn < 10) {
      const float bov = bo[nn];
#pragma unroll
      for (int r = 0; r < 8; ++r) {
        const int m = om0 + ((lane >> 4) << 3) + r;
        out[m * 10 + nn] = acc[r] * (1.0f / (float)T_STEPS) + bov;
      }
    }
  }
}

// ---------------------------------------------------------------------------
extern "C" void kernel_launch(void* const* d_in, const int* in_sizes, int n_in,
                              void* d_out, int out_size, void* d_ws, size_t ws_size,
                              hipStream_t stream) {
  const float* x    = (const float*)d_in[0];
  const float* Wi1  = (const float*)d_in[1];
  const float* bi1  = (const float*)d_in[2];
  const float* Wh1f = (const float*)d_in[3];
  const float* bh1  = (const float*)d_in[4];
  const float* Wi2f = (const float*)d_in[5];
  const float* bi2  = (const float*)d_in[6];
  const float* Wh2f = (const float*)d_in[7];
  const float* bh2  = (const float*)d_in[8];
  const float* Wi3f = (const float*)d_in[9];
  const float* bi3  = (const float*)d_in[10];
  const float* Whof = (const float*)d_in[11];
  const float* bo   = (const float*)d_in[12];
  const float* tau1 = (const float*)d_in[13];
  const float* tau2 = (const float*)d_in[14];
  const float* tau3 = (const float*)d_in[15];
  const float* mask1 = (const float*)d_in[16];
  const float* mask2 = (const float*)d_in[17];
  const float* mask3 = (const float*)d_in[18];

  // Workspace carve-up (~1.9 MB).
  __bf16* wsb = (__bf16*)d_ws;
  __bf16* Who = wsb;                              // 16*512
  __bf16* s1  = Who + (size_t)16 * H;             // 2 buffers
  __bf16* s2  = s1 + 2 * (size_t)S_ELEMS;         // 2 buffers
  __bf16* s3  = s2 + 2 * (size_t)S_ELEMS;         // 3 buffers
  unsigned* cnt = (unsigned*)(s3 + 3 * (size_t)S_ELEMS);
  unsigned* rel = cnt + 64;                       // separate cacheline

  snn_prep<<<128, 256, 0, stream>>>(Whof, Who, s1, s2, s3, cnt, rel);

  snn_main<<<NBLOCKS, TPB, 0, stream>>>(
      x, Wi1, bi1, bh1, bi2, bh2, bi3, bo, tau1, tau2, tau3,
      mask1, mask2, mask3, Wh1f, Wi2f, Wh2f, Wi3f, Who,
      s1, s2, s3, cnt, rel, (float*)d_out);
}
